// FlaxGrok1BlocKSparesTop2MLPCollection_58342835749450
// MI455X (gfx1250) — compile-verified, matrix operand order checked by reading, not run on hardware
//
#include <hip/hip_runtime.h>
#include <hip/hip_bf16.h>

typedef __attribute__((ext_vector_type(16))) __bf16         v16bf;
typedef __attribute__((ext_vector_type(8)))  float          v8f;
typedef __attribute__((ext_vector_type(8)))  unsigned short us8;

namespace {
constexpr int Bc = 4, Sc = 2048, Hc = 1024, Ic = 4096, Ec = 8, Kc = 2;
constexpr int Ttok   = Bc * Sc;      // 8192 tokens
constexpr int MAXR   = Ttok * Kc;    // 16384 worst-case rows per expert
constexpr int TILE_M = 32;
constexpr int XS_STRIDE = Hc + 8;    // pad to avoid LDS bank conflicts
constexpr int GS_STRIDE = 128 + 8;
}

__device__ __forceinline__ unsigned short f2bf(float f) {
  union { float f; unsigned u; } x; x.f = f;
  unsigned u = x.u;
  u += 0x7fffu + ((u >> 16) & 1u);   // round-to-nearest-even bf16
  return (unsigned short)(u >> 16);
}

__device__ __forceinline__ float gelu_tanh(float x) {
  const float c = 0.7978845608028654f;   // sqrt(2/pi)
  float t = tanhf(c * (x + 0.044715f * x * x * x));
  return 0.5f * x * (1.0f + t);
}

union Frag16 {
  v16bf v;
  unsigned short s[16];
  us8 h[2];
};

__global__ void zero_kernel(float4* __restrict__ out, int n4, int* __restrict__ cnt) {
  int gid = blockIdx.x * blockDim.x + threadIdx.x;
  if (gid < Ec) cnt[gid] = 0;
  int stride = gridDim.x * blockDim.x;
  for (int i = gid; i < n4; i += stride)
    out[i] = make_float4(0.f, 0.f, 0.f, 0.f);
}

__global__ void route_kernel(const float* __restrict__ rw, const int* __restrict__ sel,
                             int* __restrict__ cnt, int* __restrict__ rows,
                             float* __restrict__ wts) {
  int t = blockIdx.x * blockDim.x + threadIdx.x;
  if (t >= Ttok) return;
#pragma unroll
  for (int k = 0; k < Kc; ++k) {
    int e   = sel[t * Kc + k];
    float w = rw[t * Kc + k];
    int pos = atomicAdd(&cnt[e], 1);
    rows[e * MAXR + pos] = t;
    wts [e * MAXR + pos] = w;
  }
}

// src: [E][R][C] f32 row-major  ->  dst: [E][C][R] bf16 row-major (transpose+convert)
__global__ __launch_bounds__(256) void transpose_bf16_kernel(
    const float* __restrict__ src, unsigned short* __restrict__ dst, int R, int C)
{
  __shared__ unsigned short tile[64][64 + 8];
  const int e  = blockIdx.z;
  const int c0 = blockIdx.x * 64;
  const int r0 = blockIdx.y * 64;
  const float* s        = src + (size_t)e * R * C;
  unsigned short* d     = dst + (size_t)e * R * C;
  const int tid = threadIdx.x;

#pragma unroll
  for (int i = 0; i < 16; ++i) {
    int idx = i * 256 + tid;            // 0..4095
    int r = idx >> 6, c = idx & 63;
    tile[r][c] = f2bf(s[(size_t)(r0 + r) * C + (c0 + c)]);
  }
  __syncthreads();

#pragma unroll
  for (int i = 0; i < 4; ++i) {
    int idx4 = i * 256 + tid;           // 0..1023
    int c  = idx4 >> 4;                 // 0..63  (output row = original col)
    int k4 = (idx4 & 15) << 2;          // 0,4,...,60
    uint2 p;
    p.x = (unsigned)tile[k4 + 0][c] | ((unsigned)tile[k4 + 1][c] << 16);
    p.y = (unsigned)tile[k4 + 2][c] | ((unsigned)tile[k4 + 3][c] << 16);
    *(uint2*)(d + (size_t)(c0 + c) * R + (r0 + k4)) = p;
  }
}

__global__ __launch_bounds__(256) void moe_gemm_kernel(
    const float*          __restrict__ hs,
    const int*            __restrict__ rows,
    const float*          __restrict__ wts,
    const int*            __restrict__ cnt,
    const unsigned short* __restrict__ WtIn,   // [E][I][H] bf16, K(H)-contiguous
    const unsigned short* __restrict__ WtV,    // [E][I][H] bf16
    const unsigned short* __restrict__ WtOut,  // [E][H][I] bf16, K(I)-contiguous
    float*                __restrict__ out)
{
  const int ex        = blockIdx.y;
  const int count     = cnt[ex];
  const int tileStart = blockIdx.x * TILE_M;
  if (tileStart >= count) return;    // uniform: EXEC stays all-1s below

  __shared__ __align__(16) unsigned short Xs[TILE_M * XS_STRIDE]; // 32x1024 bf16
  __shared__ __align__(16) unsigned short Gs[TILE_M * GS_STRIDE]; // 32x128  bf16
  __shared__ int   tokIdx[TILE_M];
  __shared__ float tokW[TILE_M];

  const int tid  = threadIdx.x;
  const int wave = tid >> 5;
  const int lane = tid & 31;
  const int half = lane >> 4;
  const int nl   = lane & 15;

  if (tid < TILE_M) {
    int r = tileStart + tid;
    int tok = -1; float w = 0.0f;
    if (r < count) { tok = rows[ex * MAXR + r]; w = wts[ex * MAXR + r]; }
    tokIdx[tid] = tok;
    tokW[tid]   = w;
  }
  __syncthreads();

  // ---- stage gathered X tile into LDS as bf16 (coalesced float4 loads) ----
  for (int it = 0; it < (TILE_M * Hc / 4) / 256; ++it) {
    int idx4 = it * 256 + tid;
    int row  = idx4 >> 8;            // 256 float4 per row
    int c4   = (idx4 & 255) << 2;
    int tok  = tokIdx[row];
    float4 v = make_float4(0.f, 0.f, 0.f, 0.f);
    if (tok >= 0) v = *(const float4*)(hs + (size_t)tok * Hc + c4);
    unsigned short* dst = &Xs[row * XS_STRIDE + c4];
    dst[0] = f2bf(v.x); dst[1] = f2bf(v.y); dst[2] = f2bf(v.z); dst[3] = f2bf(v.w);
  }
  __syncthreads();

  const unsigned short* Win = WtIn  + (size_t)ex * Hc * Ic;
  const unsigned short* Wv  = WtV   + (size_t)ex * Hc * Ic;
  const unsigned short* Wo  = WtOut + (size_t)ex * Ic * Hc;

  v8f zero8 = {};
  v8f accL[8], accH[8];    // out rows 0-15 / 16-31, wave's 128 H-columns
#pragma unroll
  for (int t = 0; t < 8; ++t) { accL[t] = zero8; accH[t] = zero8; }

  for (int ic = 0; ic < Ic; ic += 128) {
    // ---- phase A: this wave's 16 columns of G = gelu(X@Win) * (X@Wv) ----
    v8f a1l = zero8, a1h = zero8, a2l = zero8, a2h = zero8;
    const int colA = ic + wave * 16 + nl;
    const unsigned short* rowIn = Win + (size_t)colA * Hc;  // K-contiguous row
    const unsigned short* rowV  = Wv  + (size_t)colA * Hc;
    for (int k0 = 0; k0 < Hc; k0 += 32) {
      Frag16 al, ah;
      const unsigned short* xsl = &Xs[nl * XS_STRIDE + k0 + half * 8];
      const unsigned short* xsh = &Xs[(nl + 16) * XS_STRIDE + k0 + half * 8];
      al.h[0] = *(const us8*)(xsl);
      al.h[1] = *(const us8*)(xsl + 16);
      ah.h[0] = *(const us8*)(xsh);
      ah.h[1] = *(const us8*)(xsh + 16);
      Frag16 bi, bv;
      const unsigned short* pin = rowIn + k0 + half * 8;
      const unsigned short* pv  = rowV  + k0 + half * 8;
      bi.h[0] = *(const us8*)(pin);
      bi.h[1] = *(const us8*)(pin + 16);
      bv.h[0] = *(const us8*)(pv);
      bv.h[1] = *(const us8*)(pv + 16);
      // B fragments reused across both row-halves: 4 WMMAs per 4 global_b128
      a1l = __builtin_amdgcn_wmma_f32_16x16x32_bf16(false, al.v, false, bi.v,
                                                    (short)0, a1l, false, false);
      a1h = __builtin_amdgcn_wmma_f32_16x16x32_bf16(false, ah.v, false, bi.v,
                                                    (short)0, a1h, false, false);
      a2l = __builtin_amdgcn_wmma_f32_16x16x32_bf16(false, al.v, false, bv.v,
                                                    (short)0, a2l, false, false);
      a2h = __builtin_amdgcn_wmma_f32_16x16x32_bf16(false, ah.v, false, bv.v,
                                                    (short)0, a2h, false, false);
    }
#pragma unroll
    for (int v = 0; v < 8; ++v) {
      int m = v + half * 8;                     // C/D layout: M = vgpr + 8*half
      Gs[m * GS_STRIDE + wave * 16 + nl]        = f2bf(gelu_tanh(a1l[v]) * a2l[v]);
      Gs[(m + 16) * GS_STRIDE + wave * 16 + nl] = f2bf(gelu_tanh(a1h[v]) * a2h[v]);
    }
    __syncthreads();

    // ---- phase B: acc(32 x 128 cols of H) += G_chunk @ W_out[ic:ic+128, :] ----
#pragma unroll
    for (int t = 0; t < 8; ++t) {
      const int nB = wave * 128 + t * 16 + nl;
      const unsigned short* rowO = Wo + (size_t)nB * Ic + ic;  // K-contiguous
#pragma unroll
      for (int k0 = 0; k0 < 128; k0 += 32) {
        Frag16 gl, gh;
        const unsigned short* gspl = &Gs[nl * GS_STRIDE + k0 + half * 8];
        const unsigned short* gsph = &Gs[(nl + 16) * GS_STRIDE + k0 + half * 8];
        gl.h[0] = *(const us8*)(gspl);
        gl.h[1] = *(const us8*)(gspl + 16);
        gh.h[0] = *(const us8*)(gsph);
        gh.h[1] = *(const us8*)(gsph + 16);
        Frag16 bo;
        const unsigned short* po = rowO + k0 + half * 8;
        bo.h[0] = *(const us8*)(po);
        bo.h[1] = *(const us8*)(po + 16);
        // B fragment reused: 2 WMMAs per 2 global_b128
        accL[t] = __builtin_amdgcn_wmma_f32_16x16x32_bf16(false, gl.v, false, bo.v,
                                                          (short)0, accL[t], false, false);
        accH[t] = __builtin_amdgcn_wmma_f32_16x16x32_bf16(false, gh.v, false, bo.v,
                                                          (short)0, accH[t], false, false);
      }
    }
    __syncthreads();
  }

  // ---- epilogue: scale by routing weight, accumulate into output ----
#pragma unroll
  for (int t = 0; t < 8; ++t) {
    const int nB = wave * 128 + t * 16 + nl;
#pragma unroll
    for (int v = 0; v < 8; ++v) {
      int mL = v + half * 8;
      int mH = mL + 16;
      int tokL = tokIdx[mL];
      int tokH = tokIdx[mH];
      if (tokL >= 0)
        unsafeAtomicAdd(out + (size_t)tokL * Hc + nB, accL[t][v] * tokW[mL]);
      if (tokH >= 0)
        unsafeAtomicAdd(out + (size_t)tokH * Hc + nB, accH[t][v] * tokW[mH]);
    }
  }
}

extern "C" void kernel_launch(void* const* d_in, const int* in_sizes, int n_in,
                              void* d_out, int out_size, void* d_ws, size_t ws_size,
                              hipStream_t stream) {
  const float* hs  = (const float*)d_in[0];  // [B,S,H]
  const float* rw  = (const float*)d_in[1];  // [B,S,K]
  const int*   sel = (const int*)  d_in[2];  // [B,S,K]
  const float* Wi  = (const float*)d_in[3];  // [E,H,I]
  const float* Wv  = (const float*)d_in[4];  // [E,H,I]
  const float* Wo  = (const float*)d_in[5];  // [E,I,H]
  float* out = (float*)d_out;

  // workspace layout
  char* ws = (char*)d_ws;
  int*   cnt    = (int*)ws;
  int*   rowsB  = (int*)(ws + 256);
  float* wtsB   = (float*)(ws + 256 + (size_t)Ec * MAXR * sizeof(int));
  size_t off = 256 + 2 * (size_t)Ec * MAXR * sizeof(int);
  off = (off + 255) & ~(size_t)255;
  const size_t wElems = (size_t)Ec * Hc * Ic;          // 33.55M per matrix
  unsigned short* WtIn  = (unsigned short*)(ws + off);
  unsigned short* WtV   = WtIn  + wElems;
  unsigned short* WtOut = WtV   + wElems;

  // zero output + counters
  zero_kernel<<<2048, 256, 0, stream>>>((float4*)out, Ttok * Hc / 4, cnt);
  // build per-expert token lists
  route_kernel<<<Ttok / 256, 256, 0, stream>>>(rw, sel, cnt, rowsB, wtsB);
  // transpose+convert weights to K-major bf16
  dim3 tgIn(Ic / 64, Hc / 64, Ec);   // src [H,I] -> dst [I,H]
  transpose_bf16_kernel<<<tgIn, 256, 0, stream>>>(Wi, WtIn, Hc, Ic);
  transpose_bf16_kernel<<<tgIn, 256, 0, stream>>>(Wv, WtV, Hc, Ic);
  dim3 tgOut(Hc / 64, Ic / 64, Ec);  // src [I,H] -> dst [H,I]
  transpose_bf16_kernel<<<tgOut, 256, 0, stream>>>(Wo, WtOut, Ic, Hc);

  // fused grouped MoE GEMM
  dim3 grid(MAXR / TILE_M, Ec);
  moe_gemm_kernel<<<grid, 256, 0, stream>>>(hs, rowsB, wtsB, cnt, WtIn, WtV, WtOut, out);
}
